// SSIMLoss_43628277792998
// MI455X (gfx1250) — matrix operand order repaired and checked by reading, not compiled, not run on hardware
//
#include <hip/hip_runtime.h>
#include <stdint.h>

typedef __attribute__((ext_vector_type(16))) __bf16       v16bf;
typedef __attribute__((ext_vector_type(2)))  __bf16       bf16x2;
typedef __attribute__((ext_vector_type(8)))  float        v8f;
typedef __attribute__((ext_vector_type(8)))  unsigned int u32x8;

#define WAVES 4   // waves per block
#define TPW   8   // tiles per wave (consecutive in x: same row, overlapping halos -> L2 hits)
#define PC 26     // product-plane col stride (u32 row-pairs); 8*26 mod 64 = 16 -> conflict-free halves
#define VS 33     // v-buffer col stride (f32); odd -> conflict-free column reads

// manual RNE f32->bf16 (used only for the 7 weights)
__device__ __forceinline__ unsigned short f2bf(float f) {
  unsigned u = __builtin_bit_cast(unsigned, f);
  u += 0x7fffu + ((u >> 16) & 1u);
  return (unsigned short)(u >> 16);
}
__device__ __forceinline__ float bf2f(unsigned short b) {
  unsigned u = ((unsigned)b) << 16;
  return __builtin_bit_cast(float, u);
}
__device__ __forceinline__ unsigned pk2(unsigned short lo, unsigned short hi) {
  return (unsigned)lo | ((unsigned)hi << 16);
}
// hardware-converted packed bf16 pair (hot path)
__device__ __forceinline__ unsigned pkbf(float a, float b) {
  bf16x2 v; v[0] = (__bf16)a; v[1] = (__bf16)b;
  return __builtin_bit_cast(unsigned, v);
}
// banded-weight select: w[d] for d in [0,6], else 0 (register-resident)
__device__ __forceinline__ unsigned short wsel(int d, const unsigned short* wb) {
  unsigned short r = 0;
#pragma unroll
  for (int j = 0; j < 7; ++j) r = (d == j) ? wb[j] : r;
  return r;
}
// separable border factor: sum of taps landing in [0,512)  (border tiles only)
__device__ __forceinline__ float edgeW(int i, const float* wfv) {
  float s = 0.f;
#pragma unroll
  for (int d = 0; d < 7; ++d) {
    int j = i + d - 3;
    s += ((unsigned)j < 512u) ? wfv[d] : 0.f;
  }
  return s;
}

__global__ void __launch_bounds__(128, 1)
ssim_wmma_kernel(const float* __restrict__ X, const float* __restrict__ Y,
                 double* __restrict__ acc) {
  __shared__ unsigned int sprod[WAVES][5][16][PC];  // bf16x2 row-pair packed, rows = K
  __shared__ float        svbuf[WAVES][16][VS];     // f32 vertical-conv intermediate

  const int lid    = threadIdx.x & 31;
  const int wv     = threadIdx.x >> 5;
  const int half   = lid >> 4;
  const int lane16 = lid & 15;
  const int waveG  = blockIdx.x * WAVES + wv;

  // ---- Gaussian window (sigma=1.5), normalized, rounded to bf16 (once per wave) ----
  float wfv[7]; unsigned short wb[7];
  {
    float raw[7]; float s = 0.f;
#pragma unroll
    for (int k = 0; k < 7; ++k) { float d = (float)(k - 3); raw[k] = expf(-d * d / 4.5f); s += raw[k]; }
#pragma unroll
    for (int k = 0; k < 7; ++k) { wb[k] = f2bf(raw[k] / s); wfv[k] = bf2f(wb[k]); }
  }
  float sumw = 0.f;
#pragma unroll
  for (int k = 0; k < 7; ++k) sumw += wfv[k];
  const float Sall = sumw * sumw;

  // ---- constant WMMA operands: banded weight matrices (once per wave) ----
  u32x8 Awv;   // vertical A: A[m,k] = w[k-m]; 16-bit A layout K = 2*(v&3) + (v&4?16:0) + 8*half
#pragma unroll
  for (int v = 0; v < 8; ++v) {
    int k0 = 2 * (v & 3) + ((v & 4) ? 16 : 0) + 8 * half;
    Awv[v] = pk2(wsel(k0 - lane16, wb), wsel(k0 + 1 - lane16, wb));
  }
  u32x8 Bwh;   // horizontal B: B[k,n] = w[k-n]; 16-bit B layout K = 2*(v&3) + (v&4?16:0) + 8*half
#pragma unroll
  for (int v = 0; v < 8; ++v) {
    int k0 = 2 * (v & 3) + ((v & 4) ? 16 : 0) + 8 * half;
    Bwh[v] = pk2(wsel(k0 - lane16, wb), wsel(k0 + 1 - lane16, wb));
  }

  // ---- zero-fill pad regions once (K rows 22..31 of product planes; vbuf cols 22..)
  // never overwritten by staging/V-stores, so they stay valid across all tiles
  for (int i = lid; i < 5 * 5 * PC; i += 32) {
    int pl = i / (5 * PC); int r = i % (5 * PC);
    sprod[wv][pl][11 + r / PC][r % PC] = 0u;
  }
  for (int i = lid; i < 16 * (VS - 22); i += 32) {
    svbuf[wv][i / (VS - 22)][22 + i % (VS - 22)] = 0.f;
  }

  const v8f cz = {0.f, 0.f, 0.f, 0.f, 0.f, 0.f, 0.f, 0.f};
  const float C1 = 1e-4f, C2 = 9e-4f;
  float accl = 0.f;

  for (int it = 0; it < TPW; ++it) {
    const int t     = waveG * TPW + it;
    const int plane = t >> 10;
    const int rem   = t & 1023;
    const int tileY = (rem >> 5) << 4;
    const int tileX = (rem & 31) << 4;
    const size_t base = (size_t)plane * (512 * 512);
    // patch spans rows/cols [tile-3, tile+18]; interior iff fully in-bounds (uniform branch)
    const bool interior = (tileX > 0) && (tileX < 496) && (tileY > 0) && (tileY < 496);

    // ---- stage shifted products: px = x - 1/2 (kills E[x^2]-mu^2 bf16 cancellation) ----
    if (interior) {
      for (int e = lid; e < 11 * 22; e += 32) {
        int c  = e % 22;
        int pr = e / 22;
        size_t a0 = base + (size_t)(tileY + 2 * pr - 3) * 512 + (tileX + c - 3);
        float px0 = X[a0]       - 0.5f, py0 = Y[a0]       - 0.5f;
        float px1 = X[a0 + 512] - 0.5f, py1 = Y[a0 + 512] - 0.5f;
        sprod[wv][0][pr][c] = pkbf(px0,       px1);
        sprod[wv][1][pr][c] = pkbf(py0,       py1);
        sprod[wv][2][pr][c] = pkbf(px0 * px0, px1 * px1);
        sprod[wv][3][pr][c] = pkbf(py0 * py0, py1 * py1);
        sprod[wv][4][pr][c] = pkbf(px0 * py0, px1 * py1);
      }
    } else {
      for (int e = lid; e < 11 * 22; e += 32) {
        int c  = e % 22;
        int pr = e / 22;
        int gc  = tileX + c - 3;
        int gr0 = tileY + 2 * pr - 3;
        bool cok = (unsigned)gc < 512u;
        bool i0  = cok && ((unsigned)gr0 < 512u);
        bool i1  = cok && ((unsigned)(gr0 + 1) < 512u);
        size_t a0 = base + (size_t)gr0 * 512 + gc;
        float x0 = i0 ? X[a0]       : 0.5f;     // OOB -> shifted value 0 (zero padding)
        float y0 = i0 ? Y[a0]       : 0.5f;
        float x1 = i1 ? X[a0 + 512] : 0.5f;
        float y1 = i1 ? Y[a0 + 512] : 0.5f;
        float px0 = x0 - 0.5f, py0 = y0 - 0.5f;
        float px1 = x1 - 0.5f, py1 = y1 - 0.5f;
        sprod[wv][0][pr][c] = pkbf(px0,       px1);
        sprod[wv][1][pr][c] = pkbf(py0,       py1);
        sprod[wv][2][pr][c] = pkbf(px0 * px0, px1 * px1);
        sprod[wv][3][pr][c] = pkbf(py0 * py0, py1 * py1);
        sprod[wv][4][pr][c] = pkbf(px0 * py0, px1 * py1);
      }
    }

    // ---- 5 planes x (2 vertical + 1 horizontal) WMMA ----
    v8f outs[5];
#pragma unroll
    for (int pl = 0; pl < 5; ++pl) {
      u32x8 b0, b1;
#pragma unroll
      for (int v = 0; v < 8; ++v) {
        int pr = (v & 3) + ((v & 4) ? 8 : 0) + 4 * half; // B layout: K=2(v&3)+(v&4?16:0)+8*half, pair-packed
        b0[v] = sprod[wv][pl][pr][lane16];          // patch cols 0..15
        b1[v] = sprod[wv][pl][pr][lane16 + 6];      // patch cols 6..21
      }
      v8f V0 = __builtin_amdgcn_wmma_f32_16x16x32_bf16(
          false, __builtin_bit_cast(v16bf, Awv), false, __builtin_bit_cast(v16bf, b0),
          (short)0, cz, false, false);
      v8f V1 = __builtin_amdgcn_wmma_f32_16x16x32_bf16(
          false, __builtin_bit_cast(v16bf, Awv), false, __builtin_bit_cast(v16bf, b1),
          (short)0, cz, false, false);
#pragma unroll
      for (int r = 0; r < 8; ++r) {                 // D-layout -> LDS transpose staging
        int m = r + 8 * half;
        svbuf[wv][m][lane16]     = V0[r];
        svbuf[wv][m][lane16 + 6] = V1[r];           // overlap cols write identical values
      }
      u32x8 av;
#pragma unroll
      for (int v = 0; v < 8; ++v) {
        int k0 = 2 * (v & 3) + ((v & 4) ? 16 : 0) + 8 * half;
        av[v] = pkbf(svbuf[wv][lane16][k0], svbuf[wv][lane16][k0 + 1]);
      }
      outs[pl] = __builtin_amdgcn_wmma_f32_16x16x32_bf16(
          false, __builtin_bit_cast(v16bf, av), false, __builtin_bit_cast(v16bf, Bwh),
          (short)0, cz, false, false);
    }

    // ---- f32 SSIM epilogue (exact un-shift; S = conv of in-bounds indicator) ----
    if (interior) {
      const float cS = 0.5f * Sall, qS = 0.25f * Sall;
#pragma unroll
      for (int r = 0; r < 8; ++r) {
        float m1p = outs[0][r], m2p = outs[1][r];
        float Pxx = outs[2][r], Pyy = outs[3][r], Pxy = outs[4][r];
        float mu1 = m1p + cS, mu2 = m2p + cS;
        float s1  = (Pxx + m1p + qS) - mu1 * mu1;
        float s2  = (Pyy + m2p + qS) - mu2 * mu2;
        float s12 = (Pxy + 0.5f * (m1p + m2p) + qS) - mu1 * mu2;
        float num = (2.f * mu1 * mu2 + C1) * (2.f * s12 + C2);
        float den = (mu1 * mu1 + mu2 * mu2 + C1) * (s1 + s2 + C2);
        accl += num / den;
      }
    } else {
      float colW = edgeW(tileX + lane16, wfv);
#pragma unroll
      for (int r = 0; r < 8; ++r) {
        int m = r + 8 * half;
        float S   = edgeW(tileY + m, wfv) * colW;
        float m1p = outs[0][r], m2p = outs[1][r];
        float Pxx = outs[2][r], Pyy = outs[3][r], Pxy = outs[4][r];
        float mu1 = m1p + 0.5f * S, mu2 = m2p + 0.5f * S;
        float s1  = (Pxx + m1p + 0.25f * S) - mu1 * mu1;
        float s2  = (Pyy + m2p + 0.25f * S) - mu2 * mu2;
        float s12 = (Pxy + 0.5f * (m1p + m2p) + 0.25f * S) - mu1 * mu2;
        float num = (2.f * mu1 * mu2 + C1) * (2.f * s12 + C2);
        float den = (mu1 * mu1 + mu2 * mu2 + C1) * (s1 + s2 + C2);
        accl += num / den;
      }
    }
  }

  // ---- one reduction + atomic per wave (amortized over TPW tiles) ----
#pragma unroll
  for (int off = 16; off > 0; off >>= 1) accl += __shfl_xor(accl, off, 32);
  if (lid == 0) atomicAdd(acc, (double)accl);
}

__global__ void ssim_finalize_kernel(const double* __restrict__ acc, float* __restrict__ out) {
  out[0] = 1.0f - (float)(acc[0] * (1.0 / 12582912.0));   // 16*3*512*512
}

extern "C" void kernel_launch(void* const* d_in, const int* in_sizes, int n_in,
                              void* d_out, int out_size, void* d_ws, size_t ws_size,
                              hipStream_t stream) {
  const float* X = (const float*)d_in[0];   // enhanced
  const float* Y = (const float*)d_in[1];   // target
  float* out = (float*)d_out;
  double* acc = (double*)d_ws;

  hipMemsetAsync(acc, 0, sizeof(double), stream);
  // 48 planes * 32*32 tiles = 49152 tiles; 4 waves/block * 8 tiles/wave -> 1536 blocks
  ssim_wmma_kernel<<<dim3(49152 / (WAVES * TPW)), dim3(128), 0, stream>>>(X, Y, acc);
  ssim_finalize_kernel<<<dim3(1), dim3(1), 0, stream>>>(acc, out);
}
